// ChimeraBlock_80934363725837
// MI455X (gfx1250) — matrix operand order repaired and checked by reading, not compiled, not run on hardware
//
#include <hip/hip_runtime.h>
#include <hip/hip_bf16.h>

// ---------------------------------------------------------------------------
// ChimeraBlock (Mamba-2 style) for MI455X / gfx1250, wave32.
// Pipeline:
//   1. f32 -> bf16 casts of u, w_in, w_out (GEMM operands)
//   2. zxBCdt = u @ w_in^T   (bf16 WMMA 16x16x32, f32 accum, TDM-fed LDS
//      double buffering when the tensor_load_to_lds builtin is available)
//   3. depthwise 3x3 conv + bias + SiLU on xBC channels (NHWC)
//   4. selective scans: one WG per (b, head, direction); both B/C groups of a
//      direction fused in one WG -> y_fwd / y_rev buffers, no atomics
//   5. gating (y * silu(z)) + RMSNorm -> bf16
//   6. out = yn @ w_out^T    (bf16 WMMA)
// ---------------------------------------------------------------------------

typedef unsigned short u16;
typedef __attribute__((ext_vector_type(16))) __bf16 v16bf;
typedef __attribute__((ext_vector_type(8)))  __bf16 v8bf;
typedef __attribute__((ext_vector_type(8)))  float  v8f;

#define D_MODEL   768
#define QK        64
#define HEADDIM   64
#define D_INNER   1536
#define NHEADS    24
#define NUM_BC    2
#define D_IN_PROJ 3424   // 2*1536 + 64*2*2 + 24*2*2
#define CONV_DIM  1792   // 1536 + 256
#define IH        32
#define IW        32
#define BSZ       32
#define SEQ       1024
#define MROWS     (BSZ * SEQ)      // 32768
#define DT_OFF    3328             // 2*D_INNER + 2*NUM_BC*QK
#define BC_OFF    1536             // inside xBC_conv row
#define CC_OFF    1664

#if defined(__gfx1250__) && __has_builtin(__builtin_amdgcn_tensor_load_to_lds) && \
    __has_builtin(__builtin_amdgcn_s_wait_tensorcnt)
#define USE_TDM 1
#else
#define USE_TDM 0
#endif

__device__ __forceinline__ u16 f2bf(float f) {
  unsigned u = __float_as_uint(f);
  u += 0x7FFFu + ((u >> 16) & 1u);       // round-to-nearest-even
  return (u16)(u >> 16);
}
__device__ __forceinline__ float siluf(float x) {
  return x / (1.f + __expf(-x));
}
__device__ __forceinline__ float softplusf(float x) {
  return (x > 20.f) ? x : log1pf(__expf(x));
}
__device__ __forceinline__ v16bf pack16(v8bf lo, v8bf hi) {
  v16bf r;
#pragma unroll
  for (int i = 0; i < 8; ++i) { r[i] = lo[i]; r[i + 8] = hi[i]; }
  return r;
}

// The TDM writes LDS through a descriptor, invisibly to clang's alias
// analysis. Escape the LDS pointer through an opaque asm with a memory
// clobber so the compiler must treat the buffer as written by the DMA
// (otherwise it folds the ds_load fragment reads into undef).
__device__ __forceinline__ void lds_written_by_dma(const void* p) {
  asm volatile("" : : "r"(p) : "memory");
}

// -------------------------------- f32 -> bf16 ------------------------------
__global__ __launch_bounds__(256) void cvt_bf16_kernel(
    const float* __restrict__ s, u16* __restrict__ d, long long n) {
  long long i = (long long)blockIdx.x * blockDim.x + threadIdx.x;
  long long stride = (long long)gridDim.x * blockDim.x;
  for (; i < n; i += stride) d[i] = f2bf(s[i]);
}

#if USE_TDM
typedef unsigned int u32x4 __attribute__((ext_vector_type(4)));
typedef int          i32x8 __attribute__((ext_vector_type(8)));
typedef int          i32x4 __attribute__((ext_vector_type(4)));

// Issue one TDM 2D tile load: tile 32 (K) x 128 (rows) of 2-byte elements
// from a row-major (rows x strideElems) tensor into LDS at byte offset
// lds_off. remCols/remRows are the remaining tensor extents measured from the
// tile origin; TDM returns zeros for reads beyond them (handles N tails).
// Descriptor layout per CDNA5 ISA ch.8 (D# groups 0/1; 2D -> groups 2,3 and
// the trailing group are zero). This toolchain uses the 6-arg builtin form.
__device__ __forceinline__ void tdm_tile_load(unsigned lds_off,
                                              const u16* gptr, int remCols,
                                              int remRows, int strideElems) {
  const unsigned long long ga = (unsigned long long)(size_t)gptr;
  u32x4 g0;
  g0[0] = 1u;                                   // count=1 valid user D#
  g0[1] = lds_off;                              // lds_addr (bytes)
  g0[2] = (unsigned)ga;                         // global_addr[31:0]
  g0[3] = (unsigned)((ga >> 32) & 0x01FFFFFFull) | 0x80000000u; // [56:32]|type=2
  i32x8 g1;
  g1[0] = (int)(1u << 16);                      // data_size=1 -> 2B elements
  g1[1] = (int)(((unsigned)remCols & 0xFFFFu) << 16);           // dim0 lo
  g1[2] = (int)((((unsigned)remCols >> 16) & 0xFFFFu) |
                (((unsigned)remRows & 0xFFFFu) << 16));         // dim0 hi|dim1 lo
  g1[3] = (int)((((unsigned)remRows >> 16) & 0xFFFFu) |
                (32u << 16));                                   // dim1 hi|tile0=32
  g1[4] = 128;                                  // tile_dim1=128, tile_dim2=0
  g1[5] = strideElems;                          // tensor_dim0_stride lo32
  g1[6] = 0;                                    // stride hi / dim1_stride lo
  g1[7] = 0;
  i32x4 z4;
  z4[0] = 0; z4[1] = 0; z4[2] = 0; z4[3] = 0;
  i32x8 z8;
#pragma unroll
  for (int i = 0; i < 8; ++i) z8[i] = 0;
  __builtin_amdgcn_tensor_load_to_lds(g0, g1, z4, z4, z8, 0);
}
#endif

// ------------------------- bf16 WMMA GEMM: C = A @ W^T ---------------------
// A: (M,K) bf16 row-major, W: (N,K) bf16 row-major, C: (M,N) f32 row-major.
// Block tile 128x128, 8 waves of 32, each wave -> 32x64 (2x4 WMMA tiles).
// LDS tile layout: sA/sW are 128 rows x 32 halves, row-major.
__device__ __forceinline__ void wmma_tiles(const u16* __restrict__ sA,
                                           const u16* __restrict__ sW,
                                           int lane, int waveM, int waveN,
                                           v8f (&acc)[2][4]) {
  // A fragments per documented 16-bit 16x32 layout:
  // lanes 0-15: K {0..7,16..23}; lanes 16-31: K {8..15,24..31}
  const int ksel = (lane >> 4) * 8;
  v16bf afrag[2];
#pragma unroll
  for (int mi = 0; mi < 2; ++mi) {
    const int row = waveM * 32 + mi * 16 + (lane & 15);
    v8bf lo = *(const v8bf*)&sA[row * 32 + ksel];
    v8bf hi = *(const v8bf*)&sA[row * 32 + ksel + 16];
    afrag[mi] = pack16(lo, hi);
  }
  // B fragments: same K order as A; lanes 0-15: K {0..7,16..23};
  // lanes 16-31: K {8..15,24..31}.
  const int kb = (lane >> 4) * 8;
#pragma unroll
  for (int ni = 0; ni < 4; ++ni) {
    const int nrow = waveN * 64 + ni * 16 + (lane & 15);
    v8bf blo = *(const v8bf*)&sW[nrow * 32 + kb];
    v8bf bhi = *(const v8bf*)&sW[nrow * 32 + kb + 16];
    v16bf bfrag = pack16(blo, bhi);
#pragma unroll
    for (int mi = 0; mi < 2; ++mi) {
      acc[mi][ni] = __builtin_amdgcn_wmma_f32_16x16x32_bf16(
          false, afrag[mi], false, bfrag, (short)0, acc[mi][ni], false, false);
    }
  }
}

__global__ __launch_bounds__(256) void gemm_bf16_wmma(
    const u16* __restrict__ A, const u16* __restrict__ W,
    float* __restrict__ C, int M, int N, int K) {
  const int t     = threadIdx.x;
  const int lane  = t & 31;
  const int wv    = t >> 5;
  const int waveM = wv & 3;            // 4 waves along M (4*32 = 128)
  const int waveN = wv >> 2;           // 2 waves along N (2*64 = 128)
  const int mBlock = blockIdx.y * 128;
  const int nBlock = blockIdx.x * 128;

  v8f acc[2][4];
#pragma unroll
  for (int i = 0; i < 2; ++i)
#pragma unroll
    for (int j = 0; j < 4; ++j)
#pragma unroll
      for (int e = 0; e < 8; ++e) acc[i][j][e] = 0.f;

#if USE_TDM
  // Double-buffered LDS fed by the Tensor Data Mover: one wave issues the
  // next tile's DMA while all 8 waves run WMMA on the current buffer.
  // smem is the only LDS object -> byte offset 0; buffer b: A at b*16384,
  // W at b*16384 + 8192.
  __shared__ __align__(128) u16 smem[2 * 8192];
  const int nIter = K >> 5;
  if (wv == 0) {
    tdm_tile_load(0u, A + (size_t)mBlock * K, K, M - mBlock, K);
    tdm_tile_load(8192u, W + (size_t)nBlock * K, K, N - nBlock, K);
  }
  for (int i = 0; i < nIter; ++i) {
    const int buf = i & 1;
    if (wv == 0) {
      if (i + 1 < nIter) {
        const int kn = (i + 1) << 5;
        const unsigned nb = (unsigned)((i + 1) & 1) * 16384u;
        tdm_tile_load(nb, A + (size_t)mBlock * K + kn, K - kn, M - mBlock, K);
        tdm_tile_load(nb + 8192u, W + (size_t)nBlock * K + kn, K - kn,
                      N - nBlock, K);
        __builtin_amdgcn_s_wait_tensorcnt((short)2);  // current tile done
      } else {
        __builtin_amdgcn_s_wait_tensorcnt((short)0);
      }
    }
    __syncthreads();                       // buffer `buf` visible to all waves
    lds_written_by_dma(smem);              // DMA wrote LDS: force real reloads
    wmma_tiles(&smem[buf * 8192], &smem[buf * 8192 + 4096], lane, waveM,
               waveN, acc);
    __syncthreads();                       // reads done; buf reusable next+1
  }
#else
  // Fallback: synchronous cooperative loads through VGPRs.
  __shared__ __align__(64) u16 sA[128 * 32];
  __shared__ __align__(64) u16 sW[128 * 32];
  const int loadRow = t >> 1;          // 0..127
  const int loadOff = (t & 1) * 16;    // 0 or 16 halves (32B chunks)
  for (int kt = 0; kt < K; kt += 32) {
    {
      const u16* src = A + (size_t)(mBlock + loadRow) * K + kt + loadOff;
      u16* dst = &sA[loadRow * 32 + loadOff];
      *(v8bf*)dst       = *(const v8bf*)src;
      *((v8bf*)dst + 1) = *((const v8bf*)src + 1);
    }
    {
      const int gn = nBlock + loadRow;
      u16* dst = &sW[loadRow * 32 + loadOff];
      if (gn < N) {
        const u16* src = W + (size_t)gn * K + kt + loadOff;
        *(v8bf*)dst       = *(const v8bf*)src;
        *((v8bf*)dst + 1) = *((const v8bf*)src + 1);
      } else {
        v8bf z;
#pragma unroll
        for (int i = 0; i < 8; ++i) z[i] = (__bf16)0.f;
        *(v8bf*)dst = z; *((v8bf*)dst + 1) = z;
      }
    }
    __syncthreads();
    wmma_tiles(sA, sW, lane, waveM, waveN, acc);
    __syncthreads();
  }
#endif

  // Epilogue per C/D layout: VGPR r -> M = r (lanes 0-15) / r+8 (lanes 16-31)
  const int rsel = (lane >> 4) * 8;
  const int ncol = lane & 15;
#pragma unroll
  for (int mi = 0; mi < 2; ++mi) {
#pragma unroll
    for (int ni = 0; ni < 4; ++ni) {
      const int grow0 = mBlock + waveM * 32 + mi * 16 + rsel;
      const int gcol  = nBlock + waveN * 64 + ni * 16 + ncol;
      if (gcol < N) {
#pragma unroll
        for (int r = 0; r < 8; ++r)
          C[(size_t)(grow0 + r) * N + gcol] = acc[mi][ni][r];
      }
    }
  }
}

// --------------------- depthwise 3x3 conv + bias + SiLU --------------------
__global__ __launch_bounds__(256) void conv_silu_kernel(
    const float* __restrict__ zxBCdt, const float* __restrict__ conv_w,
    const float* __restrict__ conv_b, float* __restrict__ out) {
  long long idx = (long long)blockIdx.x * blockDim.x + threadIdx.x;
  const long long total = (long long)MROWS * CONV_DIM;
  if (idx >= total) return;
  const int c   = (int)(idx % CONV_DIM);
  const long long pos = idx / CONV_DIM;       // b*1024 + y*32 + x
  const int wx  = (int)(pos & 31);
  const int wy  = (int)((pos >> 5) & 31);
  const long long bbase = (pos >> 10) << 10;  // b*1024

  float acc = conv_b[c];
#pragma unroll
  for (int dy = -1; dy <= 1; ++dy) {
    const int yy = wy + dy;
    if (yy < 0 || yy >= IH) continue;
#pragma unroll
    for (int dx = -1; dx <= 1; ++dx) {
      const int xx = wx + dx;
      if (xx < 0 || xx >= IW) continue;
      const float v =
          zxBCdt[(size_t)(bbase + yy * IW + xx) * D_IN_PROJ + D_INNER + c];
      acc += v * conv_w[c * 9 + (dy + 1) * 3 + (dx + 1)];
    }
  }
  out[idx] = siluf(acc);
}

// ------------------------------ selective scan -----------------------------
// grid (BSZ*NHEADS, 2 directions), block 256. Thread t owns state h[n0..n0+15][p]
// with p = t&63, n0 = (t>>6)*16, for both B/C groups of its direction.
__global__ __launch_bounds__(256) void scan_kernel(
    const float* __restrict__ zxBCdt, const float* __restrict__ xc,
    float* __restrict__ y_fwd, float* __restrict__ y_rev) {
  const int bh  = blockIdx.x;
  const int b   = bh / NHEADS;
  const int h   = bh % NHEADS;
  const int rev = blockIdx.y;
  const int t   = threadIdx.x;
  const int p   = t & 63;
  const int ng  = t >> 6;              // 0..3

  __shared__ float sB[2][64];
  __shared__ float sC[2][64];
  __shared__ float sX[64];
  __shared__ float sRed[256];

  float hst[2][16];
#pragma unroll
  for (int g = 0; g < 2; ++g)
#pragma unroll
    for (int n = 0; n < 16; ++n) hst[g][n] = 0.f;

  float* __restrict__ yd = rev ? y_rev : y_fwd;

  for (int step = 0; step < SEQ; ++step) {
    const int l = rev ? (SEQ - 1 - step) : step;
    const size_t row = (size_t)b * SEQ + l;
    const float* xrow = xc + row * CONV_DIM;

    if (t < 64) {
      sX[t]    = xrow[h * HEADDIM + t];
      sB[0][t] = xrow[BC_OFF + t];
    } else if (t < 128) {
      sB[1][t - 64] = xrow[BC_OFF + 64 + (t - 64)];
    } else if (t < 192) {
      sC[0][t - 128] = xrow[CC_OFF + (t - 128)];
    } else {
      sC[1][t - 192] = xrow[CC_OFF + 64 + (t - 192)];
    }

    const float* drow = zxBCdt + row * D_IN_PROJ + DT_OFF;
    const float dt0 = softplusf(drow[(0 * 2 + rev) * NHEADS + h]);
    const float dt1 = softplusf(drow[(1 * 2 + rev) * NHEADS + h]);
    const float a0 = __expf(-dt0);
    const float a1 = __expf(-dt1);
    __syncthreads();

    const float xp = sX[p];
    float ytot = 0.f;

    { // group 0
      float acc = 0.f;
#pragma unroll
      for (int n = 0; n < 16; ++n) {
        const int ni = ng * 16 + n;
        const float hn = a0 * hst[0][n] + dt0 * sB[0][ni] * xp;
        hst[0][n] = hn;
        acc += sC[0][ni] * hn;
      }
      sRed[t] = acc;
    }
    __syncthreads();
    if (ng == 0) ytot = sRed[p] + sRed[p + 64] + sRed[p + 128] + sRed[p + 192];
    __syncthreads();
    { // group 1
      float acc = 0.f;
#pragma unroll
      for (int n = 0; n < 16; ++n) {
        const int ni = ng * 16 + n;
        const float hn = a1 * hst[1][n] + dt1 * sB[1][ni] * xp;
        hst[1][n] = hn;
        acc += sC[1][ni] * hn;
      }
      sRed[t] = acc;
    }
    __syncthreads();
    if (ng == 0) {
      ytot += sRed[p] + sRed[p + 64] + sRed[p + 128] + sRed[p + 192];
      yd[row * D_INNER + h * HEADDIM + p] = ytot;
    }
    __syncthreads();
  }
}

// ------------------------- gating + RMSNorm -> bf16 ------------------------
__global__ __launch_bounds__(256) void gate_norm_kernel(
    const float* __restrict__ zxBCdt, const float* __restrict__ y_fwd,
    const float* __restrict__ y_rev, const float* __restrict__ norm_w,
    u16* __restrict__ yn_bf16) {
  const size_t r = blockIdx.x;
  const int t = threadIdx.x;
  const float* zrow = zxBCdt + r * D_IN_PROJ;   // z is first 1536 of the row
  const float* yf = y_fwd + r * D_INNER;
  const float* yr = y_rev + r * D_INNER;

  float yg[6];
  float ss = 0.f;
#pragma unroll
  for (int j = 0; j < 6; ++j) {
    const int c = t + j * 256;
    const float yv = yf[c] + yr[c];
    const float g  = siluf(zrow[c]);
    yg[j] = yv * g;
    ss += yg[j] * yg[j];
  }
  __shared__ float sred[256];
  sred[t] = ss;
  __syncthreads();
  for (int off = 128; off > 0; off >>= 1) {
    if (t < off) sred[t] += sred[t + off];
    __syncthreads();
  }
  const float rms = rsqrtf(sred[0] * (1.f / (float)D_INNER) + 1e-5f);
#pragma unroll
  for (int j = 0; j < 6; ++j) {
    const int c = t + j * 256;
    yn_bf16[r * D_INNER + c] = f2bf(yg[j] * rms * norm_w[c]);
  }
}

// ---------------------------------------------------------------------------
extern "C" void kernel_launch(void* const* d_in, const int* in_sizes, int n_in,
                              void* d_out, int out_size, void* d_ws,
                              size_t ws_size, hipStream_t stream) {
  const float* u      = (const float*)d_in[0];
  const float* w_in   = (const float*)d_in[1];
  const float* conv_w = (const float*)d_in[2];
  const float* conv_b = (const float*)d_in[3];
  const float* norm_w = (const float*)d_in[4];
  const float* w_out  = (const float*)d_in[5];
  float* out = (float*)d_out;

  char* ws = (char*)d_ws;
  size_t off = 0;
  auto walloc = [&](size_t bytes) -> void* {
    void* p = ws + off;
    off += (bytes + 255) & ~(size_t)255;
    return p;
  };
  u16*   u_bf     = (u16*)walloc((size_t)MROWS * D_MODEL * 2);
  u16*   win_bf   = (u16*)walloc((size_t)D_IN_PROJ * D_MODEL * 2);
  u16*   wout_bf  = (u16*)walloc((size_t)D_MODEL * D_INNER * 2);
  float* zxBCdt   = (float*)walloc((size_t)MROWS * D_IN_PROJ * 4);
  float* xBC_conv = (float*)walloc((size_t)MROWS * CONV_DIM * 4);
  float* y_f      = (float*)walloc((size_t)MROWS * D_INNER * 4);
  float* y_r      = (float*)walloc((size_t)MROWS * D_INNER * 4);
  u16*   yn_bf    = (u16*)walloc((size_t)MROWS * D_INNER * 2);
  (void)ws_size; (void)in_sizes; (void)n_in; (void)out_size;

  // 1. casts to bf16 for the WMMA GEMMs
  cvt_bf16_kernel<<<2048, 256, 0, stream>>>(u, u_bf,
      (long long)MROWS * D_MODEL);
  cvt_bf16_kernel<<<1024, 256, 0, stream>>>(w_in, win_bf,
      (long long)D_IN_PROJ * D_MODEL);
  cvt_bf16_kernel<<<512, 256, 0, stream>>>(w_out, wout_bf,
      (long long)D_MODEL * D_INNER);

  // 2. in-projection: zxBCdt = u @ w_in^T   (32768 x 3424, K=768)
  {
    dim3 grid((D_IN_PROJ + 127) / 128, MROWS / 128);
    gemm_bf16_wmma<<<grid, 256, 0, stream>>>(u_bf, win_bf, zxBCdt,
                                             MROWS, D_IN_PROJ, D_MODEL);
  }

  // 3. depthwise conv + SiLU
  {
    const long long total = (long long)MROWS * CONV_DIM;
    conv_silu_kernel<<<(unsigned)((total + 255) / 256), 256, 0, stream>>>(
        zxBCdt, conv_w, conv_b, xBC_conv);
  }

  // 4. selective scans (fwd handles k=0,2; rev handles k=1,3)
  {
    dim3 grid(BSZ * NHEADS, 2);
    scan_kernel<<<grid, 256, 0, stream>>>(zxBCdt, xBC_conv, y_f, y_r);
  }

  // 5. gate + RMSNorm -> bf16
  gate_norm_kernel<<<MROWS, 256, 0, stream>>>(zxBCdt, y_f, y_r, norm_w, yn_bf);

  // 6. out-projection: out = yn @ w_out^T   (32768 x 768, K=1536)
  {
    dim3 grid(D_MODEL / 128, MROWS / 128);
    gemm_bf16_wmma<<<grid, 256, 0, stream>>>(yn_bf, wout_bf, out,
                                             MROWS, D_MODEL, D_INNER);
  }
}
